// Block_63625645523157
// MI455X (gfx1250) — matrix operand order repaired and checked by reading, not compiled, not run on hardware
//
#include <hip/hip_runtime.h>
#include <hip/hip_bf16.h>
#include <math.h>
#include <stdint.h>

// ---------------- problem constants (fixed by the reference) ----------------
#define DIMV   768
#define HEADS  12
#define DHEAD  64
#define NTOK   31
#define BATCH  512
#define MTOT   (BATCH * NTOK)      // 15872 rows, multiple of 32
#define QKV3   (3 * DIMV)          // 2304
#define HID    (4 * DIMV)          // 3072
#define MASK_VAL (-987654321.0f)

typedef __attribute__((ext_vector_type(16))) _Float16 v16h;
typedef __attribute__((ext_vector_type(8)))  float    v8f;
typedef __attribute__((ext_vector_type(2)))  _Float16 h2;

// ---------------- WMMA helpers (CDNA5 gfx1250, wave32) ----------------------
__device__ __forceinline__ v8f wmma_f16(v16h a, v16h b, v8f c) {
  // emits v_wmma_f32_16x16x32_f16
  return __builtin_amdgcn_wmma_f32_16x16x32_f16(false, a, false, b, (short)0, c,
                                                false, false);
}

// A-matrix 16x32 f16 operand (ISA 7.12.2): lane = M + 16*((k/8)%2),
// VGPR j holds k-pair k = (j&3)*2 + 8*half + 16*(j>>2).
// rowptr = pointer to this lane's A row (contiguous K).
__device__ __forceinline__ v16h load_a_row(const _Float16* rowptr, int k0, int half) {
  v16h a;
#pragma unroll
  for (int j = 0; j < 8; ++j) {
    int k = k0 + ((j & 3) << 1) + (half << 3) + ((j >> 2) << 4);
    h2 pr = *(const h2*)(rowptr + k);
    a[2 * j]     = pr[0];
    a[2 * j + 1] = pr[1];
  }
  return a;
}

// B-matrix 32x16 f16 operand: lane = N + 16*((k/8)%2), VGPR j holds k = (j&3)*2 + 8*half + 16*(j>>2).
// rowptr = pointer to this lane's column of B, stored K-contiguous (i.e. B^T row).
__device__ __forceinline__ v16h load_b_row(const _Float16* rowptr, int k0, int half) {
  v16h b;
#pragma unroll
  for (int j = 0; j < 8; ++j) {
    int k = k0 + ((j & 3) << 1) + (half << 3) + ((j >> 2) << 4);
    h2 pr = *(const h2*)(rowptr + k);
    b[2 * j]     = pr[0];
    b[2 * j + 1] = pr[1];
  }
  return b;
}

// Async global -> LDS 16-byte copy (CDNA5, tracked by ASYNCcnt).
// Per-lane LDS dest offset in a VGPR, per-lane 64-bit global src address.
__device__ __forceinline__ void async_copy_b128(uint32_t lds_off, const void* gptr) {
  asm volatile("global_load_async_to_lds_b128 %0, %1, off"
               :: "v"(lds_off), "v"((unsigned long long)(uintptr_t)gptr)
               : "memory");
}
__device__ __forceinline__ void wait_async0() {
  asm volatile("s_wait_asynccnt 0x0" ::: "memory");
}

// ---------------- weight convert + transpose: f32 [R x C] -> f16 [C x R] ----
// Coalesced both ways via a 32x32 LDS tile (33 stride to dodge bank conflicts).
__global__ __launch_bounds__(256) void transpose_to_f16_kernel(
    const float* __restrict__ in, _Float16* __restrict__ out, int R, int C) {
  __shared__ _Float16 tile[32][33];
  int tx = threadIdx.x & 31;
  int ty = threadIdx.x >> 5;          // 0..7
  int c0 = blockIdx.x * 32;
  int r0 = blockIdx.y * 32;
#pragma unroll
  for (int i = 0; i < 32; i += 8) {
    int r = r0 + ty + i, c = c0 + tx;
    if (r < R && c < C) tile[ty + i][tx] = (_Float16)in[(size_t)r * C + c];
  }
  __syncthreads();
#pragma unroll
  for (int i = 0; i < 32; i += 8) {
    int c = c0 + ty + i, r = r0 + tx;
    if (c < C && r < R) out[(size_t)c * R + r] = tile[tx][ty + i];
  }
}

// ---------------- LayerNorm (fp32 in) -> f16 out, one block per row ---------
__global__ __launch_bounds__(256) void ln_f16_kernel(
    const float* __restrict__ x, const float* __restrict__ g,
    const float* __restrict__ b, _Float16* __restrict__ out) {
  __shared__ float sdata[256];
  int row = blockIdx.x;
  int tid = threadIdx.x;
  const float* xr = x + (size_t)row * DIMV;

  float v[3];
  float s = 0.f;
#pragma unroll
  for (int i = 0; i < 3; ++i) { v[i] = xr[tid + 256 * i]; s += v[i]; }
  sdata[tid] = s;
  __syncthreads();
#pragma unroll
  for (int off = 128; off > 0; off >>= 1) {
    if (tid < off) sdata[tid] += sdata[tid + off];
    __syncthreads();
  }
  float mean = sdata[0] * (1.0f / DIMV);
  __syncthreads();

  float vs = 0.f;
#pragma unroll
  for (int i = 0; i < 3; ++i) { float d = v[i] - mean; vs += d * d; }
  sdata[tid] = vs;
  __syncthreads();
#pragma unroll
  for (int off = 128; off > 0; off >>= 1) {
    if (tid < off) sdata[tid] += sdata[tid + off];
    __syncthreads();
  }
  float rstd = rsqrtf(sdata[0] * (1.0f / DIMV) + 1e-5f);

#pragma unroll
  for (int i = 0; i < 3; ++i) {
    int c = tid + 256 * i;
    out[(size_t)row * DIMV + c] = (_Float16)((v[i] - mean) * rstd * g[c] + b[c]);
  }
}

// ---------------- blocked WMMA GEMM with async-LDS A staging ----------------
// A: [M x K] f16 row-major.  Bt: [N x K] f16 row-major (transposed B).
// Block = 4 waves covering a 32x256 output tile; each wave owns 32x64
// (2 M-tiles x 4 N-tiles = 8 f32 accumulators).
// The 32-row A strip is shared by all 4 waves: it is DMA'd into LDS in
// double-buffered 32x128 chunks with GLOBAL_LOAD_ASYNC_TO_LDS_B128 (ASYNCcnt),
// overlapping the next chunk's copy with the current chunk's WMMAs.
// A-row LDS stride = 136 halves (272 B) -> 16 distinct banks per fragment read.
// Requires: M % 32 == 0, N % 256 == 0, K % 128 == 0 (true for all call sites).
template <bool BIAS, bool GELU_ACT, bool RES, bool OUT16>
__global__ __launch_bounds__(128) void gemm_wmma_kernel(
    const _Float16* __restrict__ A, const _Float16* __restrict__ Bt,
    const float* __restrict__ bias, const float* __restrict__ res,
    float* __restrict__ outF, _Float16* __restrict__ outH,
    int M, int N, int K) {
  constexpr int KC   = 128;           // k-chunk
  constexpr int LDA  = 136;           // padded LDS row stride (halves)
  __shared__ _Float16 As[2][32][LDA];

  int tid  = threadIdx.x;
  int lane = tid & 31;
  int wave = tid >> 5;
  int idx  = lane & 15;
  int half = lane >> 4;

  int m0 = blockIdx.x * 32;
  int n0 = (blockIdx.y * 4 + wave) * 64;

  // ---- async copier assignment: thread -> (row, 64B quarter of a 128-k row)
  int crow = tid >> 2;                // 0..31
  int cq   = tid & 3;                 // 0..3
  const _Float16* gsrc = A + (size_t)(m0 + crow) * K + cq * 32;
  uint32_t lds_base = (uint32_t)(uintptr_t)&As[0][0][0];
  uint32_t lds_thr  = lds_base + (uint32_t)crow * (LDA * 2) + (uint32_t)cq * 64;

  auto issue_chunk = [&](int kc, int buf) {
    uint32_t l0 = lds_thr + (uint32_t)buf * (32 * LDA * 2);
    const _Float16* g0 = gsrc + kc;
#pragma unroll
    for (int i = 0; i < 4; ++i)
      async_copy_b128(l0 + (uint32_t)i * 16u, g0 + i * 8);
  };

  const _Float16* brow[4];
#pragma unroll
  for (int j = 0; j < 4; ++j)
    brow[j] = Bt + (size_t)(n0 + j * 16 + idx) * K;

  const _Float16* arow0 = &As[0][idx][0];
  const _Float16* arow1 = &As[0][16 + idx][0];
  constexpr uint32_t BUFH = 32 * LDA;   // halves per buffer

  v8f acc[2][4] = {};
  issue_chunk(0, 0);
  int nchunks = K / KC;
  for (int c = 0; c < nchunks; ++c) {
    wait_async0();        // my chunk-c copies have landed in LDS
    __syncthreads();      // everyone's landed; chunk c-1 compute fully done
    if (c + 1 < nchunks) issue_chunk((c + 1) * KC, (c + 1) & 1);

    const _Float16* a0p = arow0 + (c & 1) * BUFH;
    const _Float16* a1p = arow1 + (c & 1) * BUFH;
    int kbase = c * KC;
#pragma unroll
    for (int kk = 0; kk < KC; kk += 32) {
      v16h a0 = load_a_row(a0p, kk, half);
      v16h a1 = load_a_row(a1p, kk, half);
      v16h b[4];
#pragma unroll
      for (int j = 0; j < 4; ++j) b[j] = load_b_row(brow[j], kbase + kk, half);
#pragma unroll
      for (int j = 0; j < 4; ++j) {
        acc[0][j] = wmma_f16(a0, b[j], acc[0][j]);
        acc[1][j] = wmma_f16(a1, b[j], acc[1][j]);
      }
    }
  }

#pragma unroll
  for (int i = 0; i < 2; ++i) {
#pragma unroll
    for (int j = 0; j < 4; ++j) {
      int col = n0 + j * 16 + idx;
      float bv = BIAS ? bias[col] : 0.f;
#pragma unroll
      for (int r = 0; r < 8; ++r) {
        int row = m0 + i * 16 + r + 8 * half;  // C/D layout: VGPR r -> M=r+8*half
        float v = acc[i][j][r] + bv;
        if (GELU_ACT) v = 0.5f * v * (1.0f + erff(v * 0.70710678118654752440f));
        if (RES) v += res[(size_t)row * N + col];
        if (OUT16) outH[(size_t)row * N + col] = (_Float16)v;
        else       outF[(size_t)row * N + col] = v;
      }
    }
  }
}

// ---------------- LSA attention: one wave per (batch, head) -----------------
// scores = (Q K^T) * scale, diag -> MASK_VAL, softmax over 31 keys, out = P V.
// 31 tokens padded to 32 (clamped loads; pad column zeroed after softmax).
__global__ __launch_bounds__(32) void attn_kernel(
    const _Float16* __restrict__ qkv, const float* __restrict__ scale,
    _Float16* __restrict__ outH) {
  __shared__ float    sc[32][33];
  __shared__ _Float16 p[32][32];
  __shared__ _Float16 vT[64][32];   // V^T : [dim][token]

  int bh = blockIdx.x;
  int b  = bh / HEADS;
  int h  = bh % HEADS;
  int lane = threadIdx.x;
  int idx  = lane & 15;
  int half = lane >> 4;

  const _Float16* qBase = qkv + (size_t)(b * NTOK) * QKV3 + h * DHEAD;
  const _Float16* kBase = qBase + DIMV;
  const _Float16* vBase = qBase + 2 * DIMV;

  // stage V^T into LDS (token 31 padded with zeros)
  {
    int t = lane;
    if (t < NTOK) {
      const _Float16* vr = vBase + (size_t)t * QKV3;
#pragma unroll 8
      for (int d = 0; d < DHEAD; ++d) vT[d][t] = vr[d];
    } else {
#pragma unroll 8
      for (int d = 0; d < DHEAD; ++d) vT[d][t] = (_Float16)0.f;
    }
  }
  __syncthreads();

  float sc_h = scale[h];

  // ---- scores: 2x2 tiles of 16x16, K = 64 ----
#pragma unroll
  for (int i = 0; i < 2; ++i) {
    int qrow = i * 16 + idx; if (qrow > NTOK - 1) qrow = NTOK - 1;
    const _Float16* arow = qBase + (size_t)qrow * QKV3;
#pragma unroll
    for (int j = 0; j < 2; ++j) {
      int krow = j * 16 + idx; if (krow > NTOK - 1) krow = NTOK - 1;
      const _Float16* brow = kBase + (size_t)krow * QKV3;
      v8f acc = {};
#pragma unroll
      for (int k0 = 0; k0 < DHEAD; k0 += 32) {
        v16h a  = load_a_row(arow, k0, half);
        v16h bb = load_b_row(brow, k0, half);
        acc = wmma_f16(a, bb, acc);
      }
#pragma unroll
      for (int r = 0; r < 8; ++r)
        sc[i * 16 + r + 8 * half][j * 16 + idx] = acc[r] * sc_h;
    }
  }
  __syncthreads();

  // ---- masked softmax, one row per lane (fp32) ----
  {
    int row = lane;
    if (row < NTOK) {
      float vals[NTOK];
      float mx = -3.4e38f;
#pragma unroll
      for (int j = 0; j < NTOK; ++j) {
        float vv = (j == row) ? MASK_VAL : sc[row][j];
        vals[j] = vv;
        mx = fmaxf(mx, vv);
      }
      float sum = 0.f;
#pragma unroll
      for (int j = 0; j < NTOK; ++j) { vals[j] = expf(vals[j] - mx); sum += vals[j]; }
      float inv = 1.0f / sum;
#pragma unroll
      for (int j = 0; j < NTOK; ++j) p[row][j] = (_Float16)(vals[j] * inv);
      p[row][31] = (_Float16)0.f;
    } else {
#pragma unroll
      for (int j = 0; j < 32; ++j) p[row][j] = (_Float16)0.f;
    }
  }
  __syncthreads();

  // ---- out = P (32x32) x V (32x64): 2x4 WMMA tiles, K = 32 ----
#pragma unroll
  for (int i = 0; i < 2; ++i) {
    const _Float16* arow = &p[i * 16 + idx][0];
#pragma unroll
    for (int j = 0; j < 4; ++j) {
      const _Float16* brow = &vT[j * 16 + idx][0];
      v16h a  = load_a_row(arow, 0, half);
      v16h bb = load_b_row(brow, 0, half);
      v8f acc = {};
      acc = wmma_f16(a, bb, acc);
#pragma unroll
      for (int r = 0; r < 8; ++r) {
        int row = i * 16 + r + 8 * half;
        if (row < NTOK)
          outH[(size_t)(b * NTOK + row) * DIMV + h * DHEAD + j * 16 + idx] =
              (_Float16)acc[r];
      }
    }
  }
}

// ---------------- host-side launcher ----------------------------------------
extern "C" void kernel_launch(void* const* d_in, const int* in_sizes, int n_in,
                              void* d_out, int out_size, void* d_ws, size_t ws_size,
                              hipStream_t stream) {
  const float* x     = (const float*)d_in[0];
  const float* ln1_g = (const float*)d_in[1];
  const float* ln1_b = (const float*)d_in[2];
  const float* w_qkv = (const float*)d_in[3];
  const float* scale = (const float*)d_in[4];
  const float* w_o   = (const float*)d_in[5];
  const float* b_o   = (const float*)d_in[6];
  const float* ln2_g = (const float*)d_in[7];
  const float* ln2_b = (const float*)d_in[8];
  const float* w1    = (const float*)d_in[9];
  const float* b1    = (const float*)d_in[10];
  const float* w2    = (const float*)d_in[11];
  const float* b2    = (const float*)d_in[12];
  float* out = (float*)d_out;

  // bump allocator over d_ws, 256-byte aligned
  char* base = (char*)d_ws;
  size_t off = 0;
  auto alloc = [&](size_t bytes) -> void* {
    off = (off + 255) & ~(size_t)255;
    void* pp = base + off;
    off += bytes;
    return pp;
  };

  _Float16* wqkv_t = (_Float16*)alloc((size_t)QKV3 * DIMV * 2); // [2304 x 768]
  _Float16* wo_t   = (_Float16*)alloc((size_t)DIMV * DIMV * 2); // [768 x 768]
  _Float16* w1_t   = (_Float16*)alloc((size_t)HID * DIMV * 2);  // [3072 x 768]
  _Float16* w2_t   = (_Float16*)alloc((size_t)DIMV * HID * 2);  // [768 x 3072]
  _Float16* h_h    = (_Float16*)alloc((size_t)MTOT * DIMV * 2); // LN output (reused)
  _Float16* qkv_h  = (_Float16*)alloc((size_t)MTOT * QKV3 * 2);
  _Float16* attn_h = (_Float16*)alloc((size_t)MTOT * DIMV * 2);
  float*    x1     = (float*)alloc((size_t)MTOT * DIMV * 4);
  _Float16* act_h  = (_Float16*)alloc((size_t)MTOT * HID * 2);

  dim3 tb(256);
  // weight convert + transpose (f32 [R x C] -> f16 [C x R])
  transpose_to_f16_kernel<<<dim3(QKV3 / 32, DIMV / 32), tb, 0, stream>>>(
      w_qkv, wqkv_t, DIMV, QKV3);
  transpose_to_f16_kernel<<<dim3(DIMV / 32, DIMV / 32), tb, 0, stream>>>(
      w_o, wo_t, DIMV, DIMV);
  transpose_to_f16_kernel<<<dim3(HID / 32, DIMV / 32), tb, 0, stream>>>(
      w1, w1_t, DIMV, HID);
  transpose_to_f16_kernel<<<dim3(DIMV / 32, HID / 32), tb, 0, stream>>>(
      w2, w2_t, HID, DIMV);

  // LN1
  ln_f16_kernel<<<dim3(MTOT), tb, 0, stream>>>(x, ln1_g, ln1_b, h_h);

  // QKV projection: [15872 x 768] x [768 x 2304] -> f16
  gemm_wmma_kernel<false, false, false, true>
      <<<dim3(MTOT / 32, QKV3 / 256), dim3(128), 0, stream>>>(
          h_h, wqkv_t, nullptr, nullptr, nullptr, qkv_h, MTOT, QKV3, DIMV);

  // LSA attention
  attn_kernel<<<dim3(BATCH * HEADS), dim3(32), 0, stream>>>(qkv_h, scale, attn_h);

  // output projection + bias + residual -> x1 (fp32)
  gemm_wmma_kernel<true, false, true, false>
      <<<dim3(MTOT / 32, DIMV / 256), dim3(128), 0, stream>>>(
          attn_h, wo_t, b_o, x, x1, nullptr, MTOT, DIMV, DIMV);

  // LN2
  ln_f16_kernel<<<dim3(MTOT), tb, 0, stream>>>(x1, ln2_g, ln2_b, h_h);

  // MLP up + exact GELU -> f16
  gemm_wmma_kernel<true, true, false, true>
      <<<dim3(MTOT / 32, HID / 256), dim3(128), 0, stream>>>(
          h_h, w1_t, b1, nullptr, nullptr, act_h, MTOT, HID, DIMV);

  // MLP down + bias + residual -> d_out (fp32)
  gemm_wmma_kernel<true, false, true, false>
      <<<dim3(MTOT / 32, DIMV / 256), dim3(128), 0, stream>>>(
          act_h, w2_t, b2, x1, out, nullptr, MTOT, DIMV, HID);
}
